// ReplacementPolicyHead_22806276342306
// MI455X (gfx1250) — matrix operand order repaired and checked by reading, not compiled, not run on hardware
//
#include <hip/hip_runtime.h>
#include <stdint.h>

// ---------------------------------------------------------------------------
// Types for CDNA5 WMMA (wave32): v_wmma_f32_16x16x32_f16
// ---------------------------------------------------------------------------
typedef __attribute__((ext_vector_type(16))) _Float16 v16h;
typedef __attribute__((ext_vector_type(8)))  float    v8f;
typedef __attribute__((ext_vector_type(4)))  uint32_t v4u;
typedef __attribute__((ext_vector_type(2)))  uint32_t v2u;
typedef __attribute__((ext_vector_type(4)))  float    v4f;

union Frag { v16h v; v4u u[2]; };

#define NBATCH 2048
#define SEQ    64
#define DM     768
#define DP     64
#define SCALE  0.125f

// ---- LDS layout (bytes). Pitches padded to 16B to keep ds_*_b128 aligned and
// ---- spread rows across banks (1552/4 = 388 ≡ 4 mod 64 banks).
#define XPITCH 1552                 // 768 f16 + 16B pad
#define QPITCH 144                  // 64 f16 + 16B pad
#define SPITCH 272                  // 64 f32 + 16B pad (68 floats)
#define XOFF 0
#define HOFF (XOFF + SEQ * XPITCH)  //  99328
#define QOFF (HOFF + SEQ * XPITCH)  // 198656
#define KOFF (QOFF + SEQ * QPITCH)  // 207872
#define SOFF (KOFF + SEQ * QPITCH)  // 217088
#define POFF (SOFF + SEQ * SPITCH)  // 234496
#define SMEM_BYTES (POFF + 128)     // 234624  (< 320 KB per WGP)

// ---------------------------------------------------------------------------
// f32 <-> f16 helpers
// ---------------------------------------------------------------------------
__device__ __forceinline__ unsigned short f16bits(float f) {
  _Float16 h = (_Float16)f;
  return __builtin_bit_cast(unsigned short, h);
}
__device__ __forceinline__ uint32_t f16pair(float a, float b) {
  return (uint32_t)f16bits(a) | ((uint32_t)f16bits(b) << 16);
}
__device__ __forceinline__ float f16tof(unsigned short u) {
  return (float)__builtin_bit_cast(_Float16, u);
}
__device__ __forceinline__ float mishf(float v) {
  float sp = (v > 20.0f) ? v : log1pf(__expf(v));
  return v * tanhf(sp);
}

// ---------------------------------------------------------------------------
// Prep: convert an f32 row-major [K x N] weight into f16 WMMA B-fragment order.
// Tile (kt,nt) covers K[kt*32, kt*32+32) x N[nt*16, nt*16+16).
// Per-lane: n = nt*16 + (lane&15), kbase = kt*32 + (lane>>4)*16,
// dword v holds f16 pair (kbase+2v, kbase+2v+1).  Stored [tile][lane][v].
// ---------------------------------------------------------------------------
__global__ void pack_wmma_b(const float* __restrict__ W, uint32_t* __restrict__ outw,
                            int N, int NT, int total) {
  int gid = blockIdx.x * blockDim.x + threadIdx.x;
  if (gid >= total) return;
  int v    = gid & 7;
  int lane = (gid >> 3) & 31;
  int tile = gid >> 8;
  int kt = tile / NT;
  int nt = tile - kt * NT;
  int n  = nt * 16 + (lane & 15);
  int k0 = kt * 32 + ((lane >> 4) << 4) + 2 * v;
  outw[gid] = f16pair(W[(size_t)k0 * N + n], W[(size_t)(k0 + 1) * N + n]);
}

// ---------------------------------------------------------------------------
// Fused policy head: one workgroup per batch, 8 waves.
// ---------------------------------------------------------------------------
__launch_bounds__(256, 1)
__global__ void policy_head_kernel(const float* __restrict__ x,
                                   const float* __restrict__ b1,
                                   const float* __restrict__ bq,
                                   const float* __restrict__ bk,
                                   const float* __restrict__ Wp,
                                   const float* __restrict__ bp,
                                   const int*   __restrict__ indices,
                                   const uint32_t* __restrict__ W1f,
                                   const uint32_t* __restrict__ Wqf,
                                   const uint32_t* __restrict__ Wkf,
                                   float* __restrict__ out) {
  extern __shared__ char sm[];
  const int bid  = blockIdx.x;
  const int tid  = threadIdx.x;
  const int wave = tid >> 5;
  const int lane = tid & 31;
  const int mrow = lane & 15;          // row within 16-row M tile (A layout)
  const int nloc = lane & 15;          // col within 16-col N tile (C/D layout)
  const int hiK  = lane >> 4;          // lane group selects K halves
  const int aoff = hiK << 5;           // A-fragment first byte offset (K elems*2)
  const int mbase = hiK ? 8 : 0;       // C/D: lanes 16-31 hold M rows 8..15

  // ---------------- Phase 0: stage x_b as f16 row-major in LDS --------------
  {
    const v4f* xsrc = (const v4f*)(x + (size_t)bid * SEQ * DM);
    #pragma unroll 4
    for (int i = 0; i < 48; ++i) {     // 64*768/4 = 12288 float4, 48 per thread
      int idx = tid + (i << 8);
      int row = idx / 192;             // 192 float4 per row
      int c4  = idx - row * 192;
      v4f d = xsrc[idx];
      v2u p; p.x = f16pair(d.x, d.y); p.y = f16pair(d.z, d.w);
      *(v2u*)(sm + XOFF + row * XPITCH + c4 * 8) = p;
    }
  }
  __syncthreads();

  // ---------------- Phase 1: h = mish(x @ W1 + b1)  [64 x 768] --------------
  // wave w owns N tiles [6w, 6w+6); each wave does all 4 M tiles.
  for (int nt = wave * 6; nt < wave * 6 + 6; ++nt) {
    v8f acc[4] = {};
    for (int kt = 0; kt < 24; ++kt) {
      const uint32_t* bptr = W1f + (((kt * 48 + nt) << 8) + (lane << 3));
      Frag Bf; Bf.u[0] = *(const v4u*)bptr; Bf.u[1] = *(const v4u*)(bptr + 4);
      #pragma unroll
      for (int mt = 0; mt < 4; ++mt) {
        const char* ar = sm + XOFF + (mt * 16 + mrow) * XPITCH + kt * 64 + aoff;
        Frag Af; Af.u[0] = *(const v4u*)ar; Af.u[1] = *(const v4u*)(ar + 16);
        acc[mt] = __builtin_amdgcn_wmma_f32_16x16x32_f16(
            false, Af.v, false, Bf.v, (short)0, acc[mt], false, false);
      }
    }
    const int n = nt * 16 + nloc;
    const float bias = b1[n];
    #pragma unroll
    for (int mt = 0; mt < 4; ++mt)
      #pragma unroll
      for (int r = 0; r < 8; ++r) {
        float hv = mishf(acc[mt][r] + bias);
        int m = mt * 16 + mbase + r;
        *(unsigned short*)(sm + HOFF + m * XPITCH + n * 2) = f16bits(hv);
      }
  }
  __syncthreads();

  // ---------------- Phase 2: q = h@Wq+bq, k = h@Wk+bk  [64 x 64] ------------
  // waves 0-3 -> q (mt = wave), waves 4-7 -> k (mt = wave-4); 4 N tiles each.
  {
    const int isK = wave >> 2;
    const int mt  = wave & 3;
    const uint32_t* Wf = isK ? Wkf : Wqf;
    const float* bias  = isK ? bk : bq;
    const int dstoff   = isK ? KOFF : QOFF;
    v8f acc[4] = {};
    for (int kt = 0; kt < 24; ++kt) {
      const char* ar = sm + HOFF + (mt * 16 + mrow) * XPITCH + kt * 64 + aoff;
      Frag Af; Af.u[0] = *(const v4u*)ar; Af.u[1] = *(const v4u*)(ar + 16);
      #pragma unroll
      for (int nt = 0; nt < 4; ++nt) {
        const uint32_t* bptr = Wf + (((kt * 4 + nt) << 8) + (lane << 3));
        Frag Bf; Bf.u[0] = *(const v4u*)bptr; Bf.u[1] = *(const v4u*)(bptr + 4);
        acc[nt] = __builtin_amdgcn_wmma_f32_16x16x32_f16(
            false, Af.v, false, Bf.v, (short)0, acc[nt], false, false);
      }
    }
    #pragma unroll
    for (int nt = 0; nt < 4; ++nt) {
      int n = nt * 16 + nloc;
      float bv = bias[n];
      #pragma unroll
      for (int r = 0; r < 8; ++r) {
        int m = mt * 16 + mbase + r;
        *(unsigned short*)(sm + dstoff + m * QPITCH + n * 2) = f16bits(acc[nt][r] + bv);
      }
    }
  }
  __syncthreads();

  // ---------------- Phase 3: scores = (q kT) * SCALE  [64 x 64] -------------
  // B fragment of kT needs element (d, n) = k[n][d]: K runs along k's row, so
  // it is two contiguous b128 reads from row n of the k LDS tile.
  {
    const int mt = wave & 3;
    const int ntb = (wave >> 2) * 2;
    v8f acc[2] = {};
    for (int kt = 0; kt < 2; ++kt) {
      const char* ar = sm + QOFF + (mt * 16 + mrow) * QPITCH + kt * 64 + aoff;
      Frag Af; Af.u[0] = *(const v4u*)ar; Af.u[1] = *(const v4u*)(ar + 16);
      #pragma unroll
      for (int i = 0; i < 2; ++i) {
        int nt = ntb + i;
        const char* br = sm + KOFF + (nt * 16 + nloc) * QPITCH + kt * 64 + (hiK << 5);
        Frag Bf; Bf.u[0] = *(const v4u*)br; Bf.u[1] = *(const v4u*)(br + 16);
        acc[i] = __builtin_amdgcn_wmma_f32_16x16x32_f16(
            false, Af.v, false, Bf.v, (short)0, acc[i], false, false);
      }
    }
    #pragma unroll
    for (int i = 0; i < 2; ++i) {
      int n = (ntb + i) * 16 + nloc;
      #pragma unroll
      for (int r = 0; r < 8; ++r) {
        int m = mt * 16 + mbase + r;
        *(float*)(sm + SOFF + m * SPITCH + n * 4) = acc[i][r] * SCALE;
      }
    }
  }
  __syncthreads();

  // ---------------- Phase 4: promotion head prom[8][4] = k[56:64]@Wp + bp ---
  if (tid < 32) {
    int r8 = tid >> 2, c = tid & 3;
    const unsigned short* krow = (const unsigned short*)(sm + KOFF + (56 + r8) * QPITCH);
    float s = bp[c];
    #pragma unroll 8
    for (int d = 0; d < DP; ++d) s += f16tof(krow[d]) * Wp[d * 4 + c];
    ((float*)(sm + POFF))[r8 * 4 + c] = s;
  }
  __syncthreads();

  // ---------------- Phase 5: assemble policy[4288] and gather 1858 ----------
  {
    const float* sc = (const float*)(sm + SOFF);   // pitch 68 floats
    const float* pr = (const float*)(sm + POFF);   // [8][4]
    float* ob = out + (size_t)bid * 1858;
    for (int t = tid; t < 1858; t += 256) {
      int idx = indices[t];
      float v;
      if (idx < 4096) {
        v = sc[(idx >> 6) * 68 + (idx & 63)];
      } else {
        int p = idx - 4096;          // flat over [8][24]
        int i8 = p / 24;
        int j  = p - i8 * 24;
        int jd3 = j / 3;
        v = sc[(48 + i8) * 68 + 56 + jd3] + pr[jd3 * 4 + (j - jd3 * 3)] + pr[jd3 * 4 + 3];
      }
      ob[t] = v;
    }
  }
}

// ---------------------------------------------------------------------------
// Host launch
// ---------------------------------------------------------------------------
extern "C" void kernel_launch(void* const* d_in, const int* in_sizes, int n_in,
                              void* d_out, int out_size, void* d_ws, size_t ws_size,
                              hipStream_t stream) {
  const float* x  = (const float*)d_in[0];
  const float* W1 = (const float*)d_in[1];
  const float* b1 = (const float*)d_in[2];
  const float* Wq = (const float*)d_in[3];
  const float* bq = (const float*)d_in[4];
  const float* Wk = (const float*)d_in[5];
  const float* bk = (const float*)d_in[6];
  const float* Wp = (const float*)d_in[7];
  const float* bp = (const float*)d_in[8];
  const int* indices = (const int*)d_in[9];
  float* out = (float*)d_out;

  // workspace: packed f16 B-fragment weights (deterministically rebuilt per call)
  uint32_t* w1f = (uint32_t*)d_ws;           // 24*48*256 dwords = 1152 KiB
  uint32_t* wqf = w1f + 24 * 48 * 256;       // 24*4*256 dwords  =   96 KiB
  uint32_t* wkf = wqf + 24 * 4 * 256;        // 24*4*256 dwords  =   96 KiB

  {
    int total = 24 * 48 * 256;
    pack_wmma_b<<<(total + 255) / 256, 256, 0, stream>>>(W1, w1f, DM, 48, total);
  }
  {
    int total = 24 * 4 * 256;
    pack_wmma_b<<<(total + 255) / 256, 256, 0, stream>>>(Wq, wqf, DP, 4, total);
    pack_wmma_b<<<(total + 255) / 256, 256, 0, stream>>>(Wk, wkf, DP, 4, total);
  }

  (void)hipFuncSetAttribute((const void*)policy_head_kernel,
                            hipFuncAttributeMaxDynamicSharedMemorySize, SMEM_BYTES);
  policy_head_kernel<<<NBATCH, 256, SMEM_BYTES, stream>>>(
      x, b1, bq, bk, Wp, bp, indices, w1f, wqf, wkf, out);
}
